// GatedGraphRecurrentNN_34153579938394
// MI455X (gfx1250) — matrix operand order repaired and checked by reading, not compiled, not run on hardware
//
#include <hip/hip_runtime.h>

// ---------------------------------------------------------------------------
// Gated Graph Recurrent NN on MI455X (gfx1250, wave32, WMMA + TDM)
// Dominant work: dense S (4096x4096) applications -> f16 WMMA GEMMs, f32 accum.
// S kept as f16 (33.5MB, L2-resident @192MB L2). TDM stages GEMM tiles to LDS.
// ---------------------------------------------------------------------------

#define BATCH 8
#define TT    64
#define FF    8
#define NN    4096
#define HH    32
#define GG    32
#define K1    3
#define R1    64
#define R2    8

#define KT    32      // WMMA K per step
#define LDP   40      // padded LDS row stride in f16 elems (80B, 16B aligned)

typedef __attribute__((ext_vector_type(16))) _Float16     v16h;
typedef __attribute__((ext_vector_type(8)))  _Float16     v8h;
typedef __attribute__((ext_vector_type(8)))  float        v8f;
typedef __attribute__((ext_vector_type(4)))  unsigned int v4u;
typedef __attribute__((ext_vector_type(8)))  int          v8i;
typedef __attribute__((ext_vector_type(4)))  int          v4i;

#if defined(__HIP_DEVICE_COMPILE__) && __has_builtin(__builtin_amdgcn_tensor_load_to_lds)
#define USE_TDM 1
#else
#define USE_TDM 0
#endif

// ---------------------------------------------------------------------------
// f32 -> f16 convert (grid sized to element count)
// ---------------------------------------------------------------------------
__global__ __launch_bounds__(256) void convert_h_kernel(const float* __restrict__ src,
                                                        _Float16* __restrict__ dst) {
    int idx = blockIdx.x * 256 + threadIdx.x;
    dst[idx] = (_Float16)src[idx];
}

#if USE_TDM
// ---------------------------------------------------------------------------
// TDM: DMA one 128(row) x 32(f16) tile, row stride NN elems, into LDS with
// +16B padding per 64B row => LDS row stride = 80B = LDP f16. ISA 8.3/8.4.
// 6-arg builtin form: (g0 x4, g1 x8, g2 x4, g3 x4, extra x8, cpol).
// ---------------------------------------------------------------------------
__device__ __forceinline__ void tdm_load_tile_128x32(const _Float16* gsrc, unsigned lds_off) {
    unsigned long long ga = (unsigned long long)(size_t)gsrc;
    v4u g0;
    g0[0] = 1u;                                            // count=1 (valid user D#)
    g0[1] = lds_off;                                       // LDS byte address
    g0[2] = (unsigned)(ga & 0xFFFFFFFFu);                  // global_addr[31:0]
    g0[3] = (unsigned)((ga >> 32) & 0x1FFFFFFu) | (2u << 30); // addr[56:32] | type=2
    v8i g1;
    g1[0] = (int)((1u << 16)      // data_size = 1 -> 2 bytes
                | (1u << 20)      // pad_enable
                | (3u << 22)      // pad_interval: 16 DWORDs (64B)
                | (3u << 25));    // pad_amount:   4 DWORDs (16B)
    g1[1] = (int)(((unsigned)NN & 0xFFFFu) << 16);         // tensor_dim0 lo16 @ [31:16]
    g1[2] = (int)(((unsigned)NN >> 16) | (((unsigned)NN & 0xFFFFu) << 16)); // dim0 hi | dim1 lo
    g1[3] = (int)(((unsigned)NN >> 16) | (32u << 16));     // dim1 hi | tile_dim0 = 32
    g1[4] = (int)128;                                      // tile_dim1 = 128, tile_dim2 = 0
    g1[5] = (int)NN;                                       // tensor_dim0_stride lo32
    g1[6] = 0;                                             // stride hi | dim1_stride lo
    g1[7] = 0;
    v4i gz4; gz4[0] = 0; gz4[1] = 0; gz4[2] = 0; gz4[3] = 0;   // 2D: groups 2/3 unused
    v8i gz8; for (int i = 0; i < 8; ++i) gz8[i] = 0;
    __builtin_amdgcn_tensor_load_to_lds(g0, g1, gz4, gz4, gz8, 0);
}
#endif

// ---------------------------------------------------------------------------
// WMMA tap GEMM: Out[c, n] = sum_m A[c, m] * S[n, m]   (A: M x 4096 f16)
// B operand in [n][k] LDS layout == row-major tile of S_h (no transpose).
// Block: 128 threads = 4 waves (2x2), wave tile 64x64, WG tile 128x128.
// ---------------------------------------------------------------------------
__global__ __launch_bounds__(128)
void wmma_tap_gemm(const _Float16* __restrict__ A, const _Float16* __restrict__ Sh,
                   _Float16* __restrict__ Out) {
    __shared__ _Float16 Alds[2][128][LDP];
    __shared__ _Float16 Blds[2][128][LDP];

    const int tid   = threadIdx.x;
    const int lane  = tid & 31;
    const int wv    = tid >> 5;      // wave id 0..3
    const int wr    = wv >> 1;       // wave row 0..1
    const int wc    = wv & 1;        // wave col 0..1
    const int lrow  = lane & 15;
    const int lhalf = lane >> 4;

    const int rowBase = blockIdx.y * 128;
    const int colBase = blockIdx.x * 128;

    v8f acc[4][4];
    for (int i = 0; i < 4; ++i)
        for (int j = 0; j < 4; ++j)
            for (int e = 0; e < 8; ++e) acc[i][j][e] = 0.0f;

#if USE_TDM
    auto issue = [&](int buf, int ko) {
        if (wv == 0) {
            tdm_load_tile_128x32(A  + (size_t)rowBase * NN + ko,
                                 (unsigned)(size_t)(&Alds[buf][0][0]));
            tdm_load_tile_128x32(Sh + (size_t)colBase * NN + ko,
                                 (unsigned)(size_t)(&Blds[buf][0][0]));
        }
    };
#else
    auto issue = [&](int buf, int ko) {
        for (int it = 0; it < 4; ++it) {
            int e  = tid + it * 128;          // 512 chunks of 8 f16 per panel
            int r  = e >> 2;
            int c8 = (e & 3) * 8;
            v8h vb = *(const v8h*)(Sh + (size_t)(colBase + r) * NN + ko + c8);
            *(v8h*)(&Blds[buf][r][c8]) = vb;
            v8h va = *(const v8h*)(A + (size_t)(rowBase + r) * NN + ko + c8);
            *(v8h*)(&Alds[buf][r][c8]) = va;
        }
    };
#endif

    const int KSTEPS = NN / KT;      // 128
    issue(0, 0);
#if !USE_TDM
    __syncthreads();
#endif

    for (int kk = 0; kk < KSTEPS; ++kk) {
        const int cur = kk & 1;
        if (kk + 1 < KSTEPS) issue(cur ^ 1, (kk + 1) * KT);
#if USE_TDM
        // TDM is in-order per wave: with the next buffer's pair just issued,
        // tensorcnt<=2 means the current buffer's pair has landed in LDS.
        if (wv == 0) {
            if (kk + 1 < KSTEPS) __builtin_amdgcn_s_wait_tensorcnt(2);
            else                 __builtin_amdgcn_s_wait_tensorcnt(0);
        }
        __syncthreads();
#endif

        // A frags: 16x32 f16; lane = M row; lower lanes K {0..7,16..23}, upper +8.
        v16h af[4];
        for (int i = 0; i < 4; ++i) {
            const _Float16* ap = &Alds[cur][wr * 64 + i * 16 + lrow][lhalf * 8];
            v8h lo = *(const v8h*)ap;
            v8h hi = *(const v8h*)(ap + 16);
            for (int e = 0; e < 8; ++e) { af[i][e] = lo[e]; af[i][e + 8] = hi[e]; }
        }
        // B frags: 32x16 f16 in [n][k] layout; lane = N col; lower lanes K {0..7,16..23}, upper +8.
        v16h bf[4];
        for (int j = 0; j < 4; ++j) {
            const _Float16* bp = &Blds[cur][wc * 64 + j * 16 + lrow][lhalf * 8];
            v8h lo = *(const v8h*)bp;
            v8h hi = *(const v8h*)(bp + 16);
            for (int e = 0; e < 8; ++e) { bf[j][e] = lo[e]; bf[j][e + 8] = hi[e]; }
        }

        for (int i = 0; i < 4; ++i)
            for (int j = 0; j < 4; ++j)
                acc[i][j] = __builtin_amdgcn_wmma_f32_16x16x32_f16(
                    false, af[i], false, bf[j], (short)0, acc[i][j], false, false);

        __syncthreads();
    }

    // D layout: VGPR r -> M = r + 8*lhalf within subtile; N = lrow.
    for (int i = 0; i < 4; ++i)
        for (int j = 0; j < 4; ++j) {
            int col = colBase + wc * 64 + j * 16 + lrow;
            for (int r = 0; r < 8; ++r) {
                int row = rowBase + wr * 64 + i * 16 + lhalf * 8 + r;
                Out[(size_t)row * NN + col] = (_Float16)acc[i][j][r];
            }
        }
}

// ---------------------------------------------------------------------------
// Gates + main-filter contractions in one pass over the 120-feature vectors.
// Writes: deterministic sigmoid partials part[2][BATCH][16]; av/bv (f32).
// Grid: BATCH*16 blocks (b = bx>>4, chunk = bx&15), 256 threads (one n each).
// ---------------------------------------------------------------------------
__global__ __launch_bounds__(256)
void gates_kernel(const float* __restrict__ x, const _Float16* __restrict__ zh,
                  const _Float16* __restrict__ X1, const _Float16* __restrict__ X2,
                  const _Float16* __restrict__ Z1, const _Float16* __restrict__ Z2,
                  const float* __restrict__ Ca, const float* __restrict__ Cb,
                  const float* __restrict__ bc,
                  const float* __restrict__ Da, const float* __restrict__ Db,
                  const float* __restrict__ bd,
                  const float* __restrict__ Aw, const float* __restrict__ Bw,
                  float* __restrict__ avb, float* __restrict__ bvb,
                  float* __restrict__ part, int t) {
    __shared__ float sCa[HH * K1 * FF], sDa[HH * K1 * FF], sAw[HH * K1 * FF];
    __shared__ float sCb[HH * K1 * HH], sDb[HH * K1 * HH], sBw[HH * K1 * HH];
    __shared__ float sbc[HH], sbd[HH];
    __shared__ float red[2][256];

    const int tid = threadIdx.x;
    for (int i = tid; i < HH * K1 * FF; i += 256) { sCa[i] = Ca[i]; sDa[i] = Da[i]; sAw[i] = Aw[i]; }
    for (int i = tid; i < HH * K1 * HH; i += 256) { sCb[i] = Cb[i]; sDb[i] = Db[i]; sBw[i] = Bw[i]; }
    if (tid < HH) { sbc[tid] = bc[tid]; sbd[tid] = bd[tid]; }
    __syncthreads();

    const int b = blockIdx.x >> 4;
    const int n = (blockIdx.x & 15) * 256 + tid;

    float xv[24], zv[96];
    size_t xbase = ((size_t)(b * TT + t) * FF) * NN + n;
    for (int f = 0; f < FF; ++f) {
        xv[f]      = x [xbase + (size_t)f * NN];
        xv[8 + f]  = (float)X1[xbase + (size_t)f * NN];
        xv[16 + f] = (float)X2[xbase + (size_t)f * NN];
    }
    size_t zbase = ((size_t)b * HH) * NN + n;
    for (int h = 0; h < HH; ++h) {
        zv[h]      = (float)zh[zbase + (size_t)h * NN];
        zv[32 + h] = (float)Z1[zbase + (size_t)h * NN];
        zv[64 + h] = (float)Z2[zbase + (size_t)h * NN];
    }

    float su_sum = 0.0f, sf_sum = 0.0f;
    for (int h = 0; h < HH; ++h) {
        float su = sbc[h], sf = sbd[h], av = 0.0f, bv = 0.0f;
        for (int q = 0; q < 24; ++q) {
            float xq = xv[q];
            su += sCa[h * 24 + q] * xq;
            sf += sDa[h * 24 + q] * xq;
            av += sAw[h * 24 + q] * xq;
        }
        for (int q = 0; q < 96; ++q) {
            float zq = zv[q];
            su += sCb[h * 96 + q] * zq;
            sf += sDb[h * 96 + q] * zq;
            bv += sBw[h * 96 + q] * zq;
        }
        avb[zbase + (size_t)h * NN] = av;
        bvb[zbase + (size_t)h * NN] = bv;
        su_sum += 1.0f / (1.0f + __expf(-su));
        sf_sum += 1.0f / (1.0f + __expf(-sf));
    }

    red[0][tid] = su_sum; red[1][tid] = sf_sum;
    __syncthreads();
    for (int s = 128; s > 0; s >>= 1) {
        if (tid < s) { red[0][tid] += red[0][tid + s]; red[1][tid] += red[1][tid + s]; }
        __syncthreads();
    }
    if (tid == 0) {
        part[(0 * BATCH + b) * 16 + (blockIdx.x & 15)] = red[0][0];
        part[(1 * BATCH + b) * 16 + (blockIdx.x & 15)] = red[1][0];
    }
}

// ---------------------------------------------------------------------------
// z_new = tanh(u*av + f*bv + bz): pure elementwise, fully coalesced.
// Grid: BATCH*HH*NN / 256 blocks.
// ---------------------------------------------------------------------------
__global__ __launch_bounds__(256)
void znew_kernel(_Float16* __restrict__ zh,
                 const float* __restrict__ avb, const float* __restrict__ bvb,
                 const float* __restrict__ bz, const float* __restrict__ part) {
    const int idx = blockIdx.x * 256 + threadIdx.x;
    const int b = idx >> 17;              // / (HH*NN)
    const int h = (idx >> 12) & (HH - 1); // / NN % HH

    float su = 0.0f, sf = 0.0f;
    for (int c = 0; c < 16; ++c) {
        su += part[(0 * BATCH + b) * 16 + c];
        sf += part[(1 * BATCH + b) * 16 + c];
    }
    const float inv = 1.0f / (float)(HH * NN);
    zh[idx] = (_Float16)tanhf(su * inv * avb[idx] + sf * inv * bvb[idx] + bz[h]);
}

// ---------------------------------------------------------------------------
// Output filter (tanh) + per-node MLP G -> R1 -> R2, writes d_out (f32).
// ---------------------------------------------------------------------------
__global__ __launch_bounds__(256)
void output_kernel(const _Float16* __restrict__ zh,
                   const _Float16* __restrict__ Z1, const _Float16* __restrict__ Z2,
                   const float* __restrict__ Wout, const float* __restrict__ bout,
                   const float* __restrict__ W1, const float* __restrict__ b1,
                   const float* __restrict__ W2, const float* __restrict__ b2,
                   float* __restrict__ out, int t) {
    __shared__ float sW[GG * K1 * HH];
    __shared__ float sbo[GG];
    __shared__ float sW1[R1 * GG];
    __shared__ float sb1[R1];
    __shared__ float sW2[R2 * R1];
    __shared__ float sb2[R2];

    const int tid = threadIdx.x;
    for (int i = tid; i < GG * K1 * HH; i += 256) sW[i] = Wout[i];
    for (int i = tid; i < R1 * GG; i += 256) sW1[i] = W1[i];
    for (int i = tid; i < R2 * R1; i += 256) sW2[i] = W2[i];
    if (tid < R1) sb1[tid] = b1[tid];
    if (tid < GG) sbo[tid] = bout[tid];
    if (tid < R2) sb2[tid] = b2[tid];
    __syncthreads();

    const int b = blockIdx.x >> 4;
    const int n = (blockIdx.x & 15) * 256 + tid;

    float zf[96];
    size_t zbase = ((size_t)b * HH) * NN + n;
    for (int h = 0; h < HH; ++h) {
        zf[h]      = (float)zh[zbase + (size_t)h * NN];
        zf[32 + h] = (float)Z1[zbase + (size_t)h * NN];
        zf[64 + h] = (float)Z2[zbase + (size_t)h * NN];
    }

    float gv[GG];
    for (int g = 0; g < GG; ++g) {
        float s = sbo[g];
        for (int q = 0; q < 96; ++q) s += sW[g * 96 + q] * zf[q];
        gv[g] = tanhf(s);
    }
    float r1v[R1];
    for (int j = 0; j < R1; ++j) {
        float s = sb1[j];
        for (int g = 0; g < GG; ++g) s += sW1[j * GG + g] * gv[g];
        r1v[j] = s > 0.0f ? s : 0.0f;
    }
    size_t obase = (((size_t)b * TT + t) * R2) * NN + n;
    for (int i = 0; i < R2; ++i) {
        float s = sb2[i];
        for (int j = 0; j < R1; ++j) s += sW2[i * R1 + j] * r1v[j];
        out[obase + (size_t)i * NN] = s;
    }
}

// ---------------------------------------------------------------------------
// Host-side launch sequence
// ---------------------------------------------------------------------------
extern "C" void kernel_launch(void* const* d_in, const int* in_sizes, int n_in,
                              void* d_out, int out_size, void* d_ws, size_t ws_size,
                              hipStream_t stream) {
    const float* x    = (const float*)d_in[0];
    const float* z0   = (const float*)d_in[1];
    const float* S    = (const float*)d_in[2];
    const float* Aw   = (const float*)d_in[3];
    const float* Bw   = (const float*)d_in[4];
    const float* bz   = (const float*)d_in[5];
    const float* Ca   = (const float*)d_in[6];
    const float* Cb   = (const float*)d_in[7];
    const float* bc   = (const float*)d_in[8];
    const float* Da   = (const float*)d_in[9];
    const float* Db   = (const float*)d_in[10];
    const float* bd   = (const float*)d_in[11];
    const float* Wout = (const float*)d_in[12];
    const float* bout = (const float*)d_in[13];
    const float* W1   = (const float*)d_in[14];
    const float* b1   = (const float*)d_in[15];
    const float* W2   = (const float*)d_in[16];
    const float* b2   = (const float*)d_in[17];
    float* out = (float*)d_out;

    char* ws = (char*)d_ws;
    _Float16* Sh  = (_Float16*)ws;  ws += (size_t)NN * NN * 2;                 // 33.5 MB
    _Float16* X0h = (_Float16*)ws;  ws += (size_t)BATCH * TT * FF * NN * 2;    // 33.5 MB
    _Float16* X1  = (_Float16*)ws;  ws += (size_t)BATCH * TT * FF * NN * 2;    // 33.5 MB
    _Float16* X2  = (_Float16*)ws;  ws += (size_t)BATCH * TT * FF * NN * 2;    // 33.5 MB
    _Float16* zh  = (_Float16*)ws;  ws += (size_t)BATCH * HH * NN * 2;         // 2.1 MB
    _Float16* Z1  = (_Float16*)ws;  ws += (size_t)BATCH * HH * NN * 2;         // 2.1 MB
    _Float16* Z2  = (_Float16*)ws;  ws += (size_t)BATCH * HH * NN * 2;         // 2.1 MB
    float*    avb = (float*)ws;     ws += (size_t)BATCH * HH * NN * 4;         // 4.2 MB
    float*    bvb = (float*)ws;     ws += (size_t)BATCH * HH * NN * 4;         // 4.2 MB
    float*    part = (float*)ws;    ws += 2 * BATCH * 16 * 4;

    // Setup: f32 -> f16 conversions
    convert_h_kernel<<<((size_t)NN * NN) / 256, 256, 0, stream>>>(S, Sh);
    convert_h_kernel<<<((size_t)BATCH * TT * FF * NN) / 256, 256, 0, stream>>>(x, X0h);
    convert_h_kernel<<<((size_t)BATCH * HH * NN) / 256, 256, 0, stream>>>(z0, zh);

    const dim3 gX(NN / 128, (BATCH * TT * FF) / 128);  // 32 x 32
    const dim3 gZ(NN / 128, (BATCH * HH) / 128);       // 32 x 2

    // x taps for all timesteps: X1 = S.x, X2 = S.X1
    wmma_tap_gemm<<<gX, 128, 0, stream>>>(X0h, Sh, X1);
    wmma_tap_gemm<<<gX, 128, 0, stream>>>(X1, Sh, X2);

    // taps of z0
    wmma_tap_gemm<<<gZ, 128, 0, stream>>>(zh, Sh, Z1);
    wmma_tap_gemm<<<gZ, 128, 0, stream>>>(Z1, Sh, Z2);

    for (int t = 0; t < TT; ++t) {
        gates_kernel<<<BATCH * 16, 256, 0, stream>>>(x, zh, X1, X2, Z1, Z2,
                                                     Ca, Cb, bc, Da, Db, bd,
                                                     Aw, Bw, avb, bvb, part, t);
        znew_kernel<<<(BATCH * HH * NN) / 256, 256, 0, stream>>>(zh, avb, bvb, bz, part);
        // taps of the new z (used by output(t) and by gates at t+1)
        wmma_tap_gemm<<<gZ, 128, 0, stream>>>(zh, Sh, Z1);
        wmma_tap_gemm<<<gZ, 128, 0, stream>>>(Z1, Sh, Z2);
        output_kernel<<<BATCH * 16, 256, 0, stream>>>(zh, Z1, Z2, Wout, bout,
                                                      W1, b1, W2, b2, out, t);
    }
}
